// SimpleGAT_21225728377321
// MI455X (gfx1250) — matrix operand order repaired and checked by reading, not compiled, not run on hardware
//
#include <hip/hip_runtime.h>

typedef __attribute__((ext_vector_type(16))) __bf16 v16bf;
typedef __attribute__((ext_vector_type(8)))  float  v8f;

#define LEAKY 0.2f

// ---- monotonic float<->uint encoding for atomic max on floats ----
__device__ __forceinline__ unsigned gat_encf(float x) {
  unsigned u = __float_as_uint(x);
  return (u & 0x80000000u) ? ~u : (u | 0x80000000u);
}
__device__ __forceinline__ float gat_decf(unsigned u) {
  u = (u & 0x80000000u) ? (u & 0x7FFFFFFFu) : ~u;
  return __uint_as_float(u);
}

// ---------------- elementwise helpers ----------------
__global__ void gat_fill_f32(float* __restrict__ p, float v, int n) {
  int i = blockIdx.x * blockDim.x + threadIdx.x;
  if (i < n) p[i] = v;
}

__global__ void gat_fill_enc(unsigned* __restrict__ p, float v, int n) {
  int i = blockIdx.x * blockDim.x + threadIdx.x;
  if (i < n) p[i] = gat_encf(v);
}

__global__ void gat_init_bias(float* __restrict__ out,
                              const float* __restrict__ b, int n, int mask) {
  int i = blockIdx.x * blockDim.x + threadIdx.x;
  if (i < n) out[i] = b[i & mask];
}

__global__ void gat_relu(float* __restrict__ p, int n) {
  int i = blockIdx.x * blockDim.x + threadIdx.x;
  if (i < n) p[i] = fmaxf(p[i], 0.0f);
}

// ---------------- fragment packers (f32 -> bf16 in WMMA lane order) ---------
// A fragment (16-bit A 16x32, ISA 7.12.2):
//   Ap[((mt*KK + kk)*32 + lane)*16 + j] = A[row, kidx]
//   row = mt*16 + (lane&15); khalf = lane>>4
//   kidx = kk*32 + khalf*8 + (j<8 ? j : j+8)
__global__ void gat_pack_a(const float* __restrict__ A, __bf16* __restrict__ Ap,
                           int total, int K) {
  int idx = blockIdx.x * blockDim.x + threadIdx.x;
  if (idx >= total) return;
  int j    = idx & 15;
  int lane = (idx >> 4) & 31;
  int frag = idx >> 9;            // mt*KK + kk
  int KK   = K >> 5;
  int kk   = frag % KK;
  int mt   = frag / KK;
  int row  = mt * 16 + (lane & 15);
  int khalf = lane >> 4;
  int kidx = kk * 32 + khalf * 8 + ((j < 8) ? j : j + 8);
  Ap[idx] = (__bf16)A[(size_t)row * K + kidx];
}

// B fragment (dense B 32x16): same k order as A fragment
//   Bp[((nt*KK + kk)*32 + lane)*16 + j] = B[kk*32 + khalf*8 + (j<8 ? j : j+8), nt*16 + (lane&15)]
__global__ void gat_pack_b(const float* __restrict__ B, __bf16* __restrict__ Bp,
                           int total, int K, int Nc) {
  int idx = blockIdx.x * blockDim.x + threadIdx.x;
  if (idx >= total) return;
  int j    = idx & 15;
  int lane = (idx >> 4) & 31;
  int frag = idx >> 9;            // nt*KK + kk
  int KK   = K >> 5;
  int kk   = frag % KK;
  int nt   = frag / KK;
  int krow = kk * 32 + (lane >> 4) * 8 + ((j < 8) ? j : j + 8);
  int col  = nt * 16 + (lane & 15);
  Bp[idx] = (__bf16)B[(size_t)krow * Nc + col];
}

// ---------------- WMMA GEMM on pre-packed fragments -------------------------
// one wave computes a 16 x (16*T) slab of C; K fully unrolled (KK WMMAs/tile)
template <int K, int T>
__global__ __launch_bounds__(32)
void gat_wmma_gemm(const __bf16* __restrict__ Ap, const __bf16* __restrict__ Bp,
                   float* __restrict__ C, int Nc) {
  constexpr int KK = K / 32;
  const int lane = threadIdx.x;
  const int mt   = blockIdx.y;
  const int ntb  = blockIdx.x;

  v8f acc[T] = {};

  const v16bf* ap = (const v16bf*)Ap + ((size_t)mt * KK) * 32 + lane;
#pragma unroll
  for (int kk = 0; kk < KK; ++kk) {
    v16bf a = ap[(size_t)kk * 32];      // 32 bytes contiguous -> 2x b128
#pragma unroll
    for (int t = 0; t < T; ++t) {
      const v16bf* bp = (const v16bf*)Bp +
                        ((size_t)(ntb * T + t) * KK + kk) * 32 + lane;
      v16bf b = *bp;                    // 32 bytes contiguous -> 2x b128
      acc[t] = __builtin_amdgcn_wmma_f32_16x16x32_bf16(
          false, a, false, b, (short)0, acc[t], false, false);
    }
  }

  const int mrow = mt * 16 + (lane >> 4) * 8;
  const int colb = lane & 15;
#pragma unroll
  for (int t = 0; t < T; ++t) {
    const int col = (ntb * T + t) * 16 + colb;
#pragma unroll
    for (int g = 0; g < 8; ++g)
      C[(size_t)(mrow + g) * Nc + col] = acc[t][g];
  }
}

// ---------------- per-node attention dot products ----------------
__global__ __launch_bounds__(256)
void gat_dots(const float* __restrict__ h, const float* __restrict__ asv,
              const float* __restrict__ adv, float* __restrict__ ls,
              float* __restrict__ ld, int N, int D) {
  int node = blockIdx.x * (blockDim.x >> 5) + (threadIdx.x >> 5);
  int lane = threadIdx.x & 31;
  if (node >= N) return;
  const float4* hr = (const float4*)(h + (size_t)node * D);
  const float4* a1 = (const float4*)asv;
  const float4* a2 = (const float4*)adv;
  float s1 = 0.f, s2 = 0.f;
  for (int j = lane; j < (D >> 2); j += 32) {
    float4 v = hr[j], u = a1[j], w = a2[j];
    s1 += v.x * u.x + v.y * u.y + v.z * u.z + v.w * u.w;
    s2 += v.x * w.x + v.y * w.y + v.z * w.z + v.w * w.w;
  }
#pragma unroll
  for (int m = 16; m; m >>= 1) {
    s1 += __shfl_xor(s1, m, 32);
    s2 += __shfl_xor(s2, m, 32);
  }
  if (lane == 0) { ls[node] = s1; ld[node] = s2; }
}

// ---------------- edge pass 1: leaky-relu logit + segment max ---------------
__global__ void gat_logit_max(const int* __restrict__ ei,
                              const float* __restrict__ ls,
                              const float* __restrict__ ld,
                              float* __restrict__ elog,
                              unsigned* __restrict__ lmax, int E, int Etot) {
  int g = blockIdx.x * blockDim.x + threadIdx.x;
  if (g >= Etot) return;
  int s, d;
  if (g < E) { s = ei[g]; d = ei[E + g]; } else { s = d = g - E; }
  float t = ls[s] + ld[d];
  t = (t > 0.f) ? t : LEAKY * t;
  elog[g] = t;
  atomicMax(&lmax[d], gat_encf(t));
}

// ---------------- edge pass 2: exp + segment sum ----------------------------
__global__ void gat_expsum(const int* __restrict__ ei,
                           float* __restrict__ elog,
                           const unsigned* __restrict__ lmax,
                           float* __restrict__ denom, int E, int Etot) {
  int g = blockIdx.x * blockDim.x + threadIdx.x;
  if (g >= Etot) return;
  int d = (g < E) ? ei[E + g] : (g - E);
  float w = __expf(elog[g] - gat_decf(lmax[d]));
  elog[g] = w;
  atomicAdd(&denom[d], w);
}

// ---------------- edge pass 3: weighted scatter-add (one wave per edge) -----
__global__ __launch_bounds__(256)
void gat_scatter(const int* __restrict__ ei, const float* __restrict__ w,
                 const float* __restrict__ denom, const float* __restrict__ h,
                 float* __restrict__ agg, int E, int Etot, int D) {
  int wid  = blockIdx.x * (blockDim.x >> 5) + (threadIdx.x >> 5);
  int lane = threadIdx.x & 31;
  if (wid >= Etot) return;
  int s, d;
  if (wid < E) { s = ei[wid]; d = ei[E + wid]; } else { s = d = wid - E; }
  const float4* hs = (const float4*)(h + (size_t)s * D);
  __builtin_prefetch(hs, 0, 0);                    // global_prefetch_b8
  float alpha = w[wid] / denom[d];
  float* ad = agg + (size_t)d * D;
  for (int j = lane; j < (D >> 2); j += 32) {
    float4 v = hs[j];
    atomicAdd(&ad[4 * j + 0], alpha * v.x);
    atomicAdd(&ad[4 * j + 1], alpha * v.y);
    atomicAdd(&ad[4 * j + 2], alpha * v.z);
    atomicAdd(&ad[4 * j + 3], alpha * v.w);
  }
}

// ---------------------------------------------------------------------------
static inline int cdiv(int a, int b) { return (a + b - 1) / b; }

extern "C" void kernel_launch(void* const* d_in, const int* in_sizes, int n_in,
                              void* d_out, int out_size, void* d_ws, size_t ws_size,
                              hipStream_t stream) {
  const float* x   = (const float*)d_in[0];
  const int*   ei  = (const int*)d_in[1];
  const float* W1  = (const float*)d_in[2];
  const float* as1 = (const float*)d_in[3];
  const float* ad1 = (const float*)d_in[4];
  const float* b1  = (const float*)d_in[5];
  const float* W2  = (const float*)d_in[6];
  const float* as2 = (const float*)d_in[7];
  const float* ad2 = (const float*)d_in[8];
  const float* b2  = (const float*)d_in[9];

  const int DIN = 128, DMID = 256, DOUT = 128;
  const int N    = in_sizes[0] / DIN;   // 50000 (multiple of 16)
  const int E    = in_sizes[1] / 2;     // 800000
  const int Etot = E + N;               // + self loops

  // ---- workspace carve (256B aligned) ----
  char*  base = (char*)d_ws;
  size_t off  = 0;
  auto alloc = [&](size_t bytes) -> void* {
    void* p = base + off;
    off = (off + bytes + 255) & ~(size_t)255;
    return p;
  };
  __bf16*   apk1  = (__bf16*)  alloc((size_t)N * DIN * 2);
  __bf16*   bpk1  = (__bf16*)  alloc((size_t)DIN * DMID * 2);
  float*    h1    = (float*)   alloc((size_t)N * DMID * 4);
  float*    agg1  = (float*)   alloc((size_t)N * DMID * 4);
  __bf16*   apk2  = (__bf16*)  alloc((size_t)N * DMID * 2);
  __bf16*   bpk2  = (__bf16*)  alloc((size_t)DMID * DOUT * 2);
  float*    h2    = (float*)   alloc((size_t)N * DOUT * 4);
  float*    ls    = (float*)   alloc((size_t)N * 4);
  float*    ldv   = (float*)   alloc((size_t)N * 4);
  unsigned* lmax  = (unsigned*)alloc((size_t)N * 4);
  float*    denom = (float*)   alloc((size_t)N * 4);
  float*    elog  = (float*)   alloc((size_t)Etot * 4);
  (void)ws_size;

  float* out = (float*)d_out;
  const int TB = 256;
  const float NEG_BIG = -3.0e38f;

  // ================= layer 1: h1 = x @ W1 =================
  gat_pack_a<<<cdiv(N * DIN, TB), TB, 0, stream>>>(x, apk1, N * DIN, DIN);
  gat_pack_b<<<cdiv(DIN * DMID, TB), TB, 0, stream>>>(W1, bpk1, DIN * DMID, DIN, DMID);
  gat_wmma_gemm<128, 4><<<dim3(DMID / 64, N / 16), 32, 0, stream>>>(apk1, bpk1, h1, DMID);

  gat_dots<<<cdiv(N, 8), TB, 0, stream>>>(h1, as1, ad1, ls, ldv, N, DMID);

  gat_fill_enc<<<cdiv(N, TB), TB, 0, stream>>>(lmax, NEG_BIG, N);
  gat_fill_f32<<<cdiv(N, TB), TB, 0, stream>>>(denom, 0.0f, N);
  gat_init_bias<<<cdiv(N * DMID, TB), TB, 0, stream>>>(agg1, b1, N * DMID, DMID - 1);

  gat_logit_max<<<cdiv(Etot, TB), TB, 0, stream>>>(ei, ls, ldv, elog, lmax, E, Etot);
  gat_expsum<<<cdiv(Etot, TB), TB, 0, stream>>>(ei, elog, lmax, denom, E, Etot);
  gat_scatter<<<cdiv(Etot, 8), TB, 0, stream>>>(ei, elog, denom, h1, agg1, E, Etot, DMID);

  gat_relu<<<cdiv(N * DMID, TB), TB, 0, stream>>>(agg1, N * DMID);

  // ================= layer 2: h2 = relu(agg1) @ W2 =================
  gat_pack_a<<<cdiv(N * DMID, TB), TB, 0, stream>>>(agg1, apk2, N * DMID, DMID);
  gat_pack_b<<<cdiv(DMID * DOUT, TB), TB, 0, stream>>>(W2, bpk2, DMID * DOUT, DMID, DOUT);
  gat_wmma_gemm<256, 4><<<dim3(DOUT / 64, N / 16), 32, 0, stream>>>(apk2, bpk2, h2, DOUT);

  gat_dots<<<cdiv(N, 8), TB, 0, stream>>>(h2, as2, ad2, ls, ldv, N, DOUT);

  gat_fill_enc<<<cdiv(N, TB), TB, 0, stream>>>(lmax, NEG_BIG, N);
  gat_fill_f32<<<cdiv(N, TB), TB, 0, stream>>>(denom, 0.0f, N);
  gat_init_bias<<<cdiv(N * DOUT, TB), TB, 0, stream>>>(out, b2, N * DOUT, DOUT - 1);

  gat_logit_max<<<cdiv(Etot, TB), TB, 0, stream>>>(ei, ls, ldv, elog, lmax, E, Etot);
  gat_expsum<<<cdiv(Etot, TB), TB, 0, stream>>>(ei, elog, lmax, denom, E, Etot);
  gat_scatter<<<cdiv(Etot, 8), TB, 0, stream>>>(ei, elog, denom, h2, out, E, Etot, DOUT);

  gat_relu<<<cdiv(N * DOUT, TB), TB, 0, stream>>>(out, N * DOUT);
}
